// MultiHeadSelfAttention_31224412242050
// MI455X (gfx1250) — matrix operand order repaired and checked by reading, not compiled, not run on hardware
//
#include <hip/hip_runtime.h>
#include <hip/hip_bf16.h>

#define BS 4
#define SEQ 1024
#define DIM 1024
#define NH 16
#define DH 64

typedef __attribute__((ext_vector_type(16))) __bf16 v16bf;
typedef __attribute__((ext_vector_type(8)))  __bf16 v8bf;
typedef __attribute__((ext_vector_type(4)))  __bf16 v4bf;
typedef __attribute__((ext_vector_type(8)))  float  v8f;
typedef __attribute__((ext_vector_type(4)))  unsigned int v4u;
typedef __attribute__((ext_vector_type(8)))  int v8i;
typedef __attribute__((ext_vector_type(4)))  int v4i;

// round-to-nearest-even f32 -> bf16
__device__ __forceinline__ __bf16 f2bf(float f) {
  unsigned int x = __builtin_bit_cast(unsigned int, f);
  x += 0x7FFFu + ((x >> 16) & 1u);
  unsigned short h = (unsigned short)(x >> 16);
  return __builtin_bit_cast(__bf16, h);
}

__device__ __forceinline__ v16bf cat16(v8bf lo, v8bf hi) {
  return __builtin_shufflevector(lo, hi, 0, 1, 2, 3, 4, 5, 6, 7,
                                 8, 9, 10, 11, 12, 13, 14, 15);
}

__device__ __forceinline__ v8f wmma_bf16(v16bf a, v16bf b, v8f c) {
  // (neg_a, A, neg_b, B, c_mod, C, reuse_a, reuse_b)
  return __builtin_amdgcn_wmma_f32_16x16x32_bf16(false, a, false, b,
                                                 (short)0, c, false, false);
}

// ---------------------------------------------------------------------------
// Tensor Data Mover: 2D bf16 tile (tile_d0 x tile_d1 elements) from global
// into LDS at lds_addr (linear row-major).  D# bitfields per CDNA5 ISA §8.
// ---------------------------------------------------------------------------
__device__ __forceinline__ void tdm_load_2d_bf16(
    unsigned int lds_addr, const void* gaddr, unsigned int tile_d0,
    unsigned int tile_d1, unsigned int tensor_d0, unsigned int tensor_d1,
    unsigned int stride0) {
  const unsigned long long ga = (unsigned long long)(uintptr_t)gaddr;
  v4u g0;
  g0[0] = 1u;                                   // count=1 (valid), no gather
  g0[1] = lds_addr;                             // lds_addr [63:32]
  g0[2] = (unsigned int)(ga & 0xFFFFFFFFu);     // global_addr[31:0]
  g0[3] = (unsigned int)((ga >> 32) & 0x1FFFFFFu)  // global_addr[56:32]
          | (2u << 30);                         // type=2 ("image")
  v8i g1;
  g1[0] = (int)(1u << 16);                      // data_size=1 -> 2 bytes
  g1[1] = (int)((tensor_d0 & 0xFFFFu) << 16);   // tensor_dim0[15:0] @ [63:48]
  g1[2] = (int)((tensor_d0 >> 16) | ((tensor_d1 & 0xFFFFu) << 16));
  g1[3] = (int)((tensor_d1 >> 16) | (tile_d0 << 16));  // tile_dim0 @ [127:112]
  g1[4] = (int)tile_d1;                         // tile_dim1; tile_dim2=0
  g1[5] = (int)stride0;                         // tensor_dim0_stride[31:0]
  g1[6] = 0;
  g1[7] = 0;
  const v4i gz = {0, 0, 0, 0};
#if defined(__clang_major__) && (__clang_major__ >= 23)
  const v8i gz8 = {0, 0, 0, 0, 0, 0, 0, 0};
  __builtin_amdgcn_tensor_load_to_lds(g0, g1, gz, gz, gz8, 0);
#else
  __builtin_amdgcn_tensor_load_to_lds(g0, g1, gz, gz, 0);
#endif
}

// ---------------------------------------------------------------------------
// GEMM: out[M,N] = A[M,K] @ W[K,N] (+bias)*scale.  M=4096, N=K=1024.
// Block: 256 threads (8 waves), tile 128x128, K-step 32.
// A_BF16: A is bf16 (context) -> A tile moved by TDM; else f32 VALU-converted.
// VTRANS: scatter output as V^T per head: [b][h][d][s].
// ---------------------------------------------------------------------------
template <bool A_BF16, bool OUT_BF16, bool VTRANS>
__global__ __launch_bounds__(256) void mha_gemm_kernel(
    const void* __restrict__ Av, const float* __restrict__ W,
    const float* __restrict__ bias, void* __restrict__ outv, float scale) {
  constexpr int M = BS * SEQ, N = DIM, K = DIM;
  __shared__ __bf16 lA[128 * 32];   // [m][k] row-major
  __shared__ __bf16 lB[128 * 32];   // [n][k] (transposed at fill time)

  const int t = threadIdx.x;
  const int lane = t & 31;
  const int w = t >> 5;             // 0..7
  const int hl = lane >> 4;         // lane half
  const int l15 = lane & 15;
  const int m0 = blockIdx.y * 128;
  const int n0 = blockIdx.x * 128;
  const int mw = (w >> 1) * 32;     // wave's 32 rows within tile
  const int nw = (w & 1) * 64;      // wave's 64 cols within tile

  v8f acc[2][4];
#pragma unroll
  for (int mi = 0; mi < 2; ++mi)
#pragma unroll
    for (int ni = 0; ni < 4; ++ni) acc[mi][ni] = v8f{};

  for (int k0 = 0; k0 < K; k0 += 32) {
    // ---- fill A tile (128x32 bf16) ----
    if constexpr (A_BF16) {
      // Tensor Data Mover does the whole 8KB tile; wave 0 issues it.
      if (t < 32) {
        tdm_load_2d_bf16((unsigned int)(uintptr_t)&lA[0],
                         (const __bf16*)Av + (size_t)m0 * K + k0,
                         /*tile*/ 32u, 128u, /*tensor*/ (unsigned)K,
                         (unsigned)M, (unsigned)K);
      }
    } else {
      const float* A = (const float*)Av;
#pragma unroll
      for (int i = 0; i < 4; ++i) {
        int id = t + i * 256;                  // 1024 chunks of float4
        int row = id >> 3, c4 = (id & 7) * 4;
        float4 v = *(const float4*)(A + (size_t)(m0 + row) * K + k0 + c4);
        v4bf bv;
        bv.x = f2bf(v.x); bv.y = f2bf(v.y); bv.z = f2bf(v.z); bv.w = f2bf(v.w);
        *(v4bf*)&lA[row * 32 + c4] = bv;
      }
    }
    // ---- fill B tile (32x128 f32 -> [n][k] bf16) ----
#pragma unroll
    for (int i = 0; i < 4; ++i) {
      int id = t + i * 256;                    // 1024 chunks of float4
      int row = id >> 5;                       // k: 0..31
      int c4 = (id & 31) * 4;                  // n: 0..127
      float4 v = *(const float4*)(W + (size_t)(k0 + row) * N + n0 + c4);
      lB[(c4 + 0) * 32 + row] = f2bf(v.x);
      lB[(c4 + 1) * 32 + row] = f2bf(v.y);
      lB[(c4 + 2) * 32 + row] = f2bf(v.z);
      lB[(c4 + 3) * 32 + row] = f2bf(v.w);
    }
    if (k0 + 32 < K)   // hint next weight tile into cache
      __builtin_prefetch(W + (size_t)(k0 + 32) * N + n0 + (t & 31) * 4, 0, 3);
    if constexpr (A_BF16) {
      if (t < 32) __builtin_amdgcn_s_wait_tensorcnt(0);  // TDM tile landed
    }
    __syncthreads();

    // ---- fragments + WMMA ----
    v16bf aF[2], bF[4];
#pragma unroll
    for (int mi = 0; mi < 2; ++mi) {
      // A layout: lane holds row l15; lanes 0-15 K{0..7,16..23}, 16-31 K{8..15,24..31}
      const __bf16* p = &lA[(mw + mi * 16 + l15) * 32 + hl * 8];
      aF[mi] = cat16(*(const v8bf*)p, *(const v8bf*)(p + 16));
    }
#pragma unroll
    for (int ni = 0; ni < 4; ++ni) {
      // B layout: lane holds col l15; lanes 0-15 K{0..7,16..23}, 16-31 K{8..15,24..31}
      const __bf16* p = &lB[(nw + ni * 16 + l15) * 32 + hl * 8];
      bF[ni] = cat16(*(const v8bf*)p, *(const v8bf*)(p + 16));
    }
#pragma unroll
    for (int mi = 0; mi < 2; ++mi)
#pragma unroll
      for (int ni = 0; ni < 4; ++ni)
        acc[mi][ni] = wmma_bf16(aF[mi], bF[ni], acc[mi][ni]);
    __syncthreads();
  }

  // ---- epilogue: C layout is (row = r + 8*hl, col = l15) per 16x16 frag ----
#pragma unroll
  for (int mi = 0; mi < 2; ++mi) {
#pragma unroll
    for (int ni = 0; ni < 4; ++ni) {
      const int n = n0 + nw + ni * 16 + l15;
      const float bn = bias[n];
#pragma unroll
      for (int r = 0; r < 8; ++r) {
        const int m = m0 + mw + mi * 16 + hl * 8 + r;
        const float val = (acc[mi][ni][r] + bn) * scale;
        if constexpr (OUT_BF16) {
          __bf16* out = (__bf16*)outv;
          size_t idx;
          if constexpr (VTRANS) {
            const int b = m >> 10, s = m & (SEQ - 1);
            const int h = n >> 6, d = n & 63;
            idx = ((size_t)((b * NH + h) * DH + d)) * SEQ + s;
          } else {
            idx = (size_t)m * N + n;
          }
          out[idx] = f2bf(val);
        } else {
          ((float*)outv)[(size_t)m * N + n] = val;
        }
      }
    }
  }
}

// ---------------------------------------------------------------------------
// Flash attention with fused gauss-reweight:
//   weights = exp(s)*gw / sum(exp(s)*gw)   (softmax normalizer cancels)
// Block: 128 threads (4 waves); each wave owns a 16-row Q tile of one (b,h).
// Streams 32 keys/iter: 4 score WMMAs + online softmax + 4 PV WMMAs.
// Mask handled as additive -1e30 bias; running max starts at -3e38 so all
// exp() paths are finite -> no compare/select chains.
// ---------------------------------------------------------------------------
__global__ __launch_bounds__(128) void mha_flash_attn_kernel(
    const unsigned short* __restrict__ Qws, const unsigned short* __restrict__ Kws,
    const unsigned short* __restrict__ Vtws, const int* __restrict__ mask,
    const float* __restrict__ gaussw, unsigned short* __restrict__ Cws) {
  const int t = threadIdx.x;
  const int lane = t & 31;
  const int w = t >> 5;            // 0..3
  const int hl = lane >> 4;
  const int l15 = lane & 15;
  const int b = blockIdx.z;
  const int h = blockIdx.y;
  const int q0 = (blockIdx.x * 4 + w) * 16;

  __shared__ __bf16 lP[4][16 * 32];   // per-wave P re-swizzle buffer

  const __bf16* Q = (const __bf16*)Qws;
  const __bf16* Kp = (const __bf16*)Kws;
  const __bf16* Vt = (const __bf16*)Vtws;

  // Q A-fragments: rows q0..q0+15, d-head split into two K=32 chunks
  v16bf aQ[2];
  {
    const __bf16* qp = Q + (size_t)(b * SEQ + q0 + l15) * DIM + h * DH;
#pragma unroll
    for (int c = 0; c < 2; ++c) {
      const __bf16* p = qp + c * 32 + hl * 8;
      aQ[c] = cat16(*(const v8bf*)p, *(const v8bf*)(p + 16));
    }
  }

  v8f o[4] = {};                   // 16 x 64 context accumulator
  float mrow[8], lrow[8];
#pragma unroll
  for (int r = 0; r < 8; ++r) { mrow[r] = -3.0e38f; lrow[r] = 0.0f; }

  const float* gw = gaussw + b * SEQ;
  const int* mk = mask + b * SEQ;
  const __bf16* Kbase = Kp + (size_t)b * SEQ * DIM + h * DH;
  const __bf16* Vbase = Vt + (size_t)(b * NH + h) * DH * SEQ;
  __bf16* P = lP[w];

  for (int kt = 0; kt < SEQ; kt += 32) {
    // ---- scores: S = Qhat . K^T for 32 keys (two 16-col tiles) ----
    v8f sc[2] = {};
#pragma unroll
    for (int nt = 0; nt < 2; ++nt) {
      const __bf16* kp = Kbase + (size_t)(kt + nt * 16 + l15) * DIM;
#pragma unroll
      for (int c = 0; c < 2; ++c) {
        const __bf16* p = kp + c * 32 + hl * 8;
        v16bf bK = cat16(*(const v8bf*)p, *(const v8bf*)(p + 16));
        sc[nt] = wmma_bf16(aQ[c], bK, sc[nt]);
      }
    }

    // per-column mask bias + gauss weight (column = l15 within each sub-tile)
    const float g0 = gw[kt + l15] + 1e-10f;
    const float g1 = gw[kt + 16 + l15] + 1e-10f;
    const float mb0 = (mk[kt + l15] == 0) ? -1.0e30f : 0.0f;
    const float mb1 = (mk[kt + 16 + l15] == 0) ? -1.0e30f : 0.0f;

    float alpha[8], p0[8], p1[8];
#pragma unroll
    for (int r = 0; r < 8; ++r) {
      const float s0 = sc[0][r] + mb0;
      const float s1 = sc[1][r] + mb1;
      float mx = fmaxf(s0, s1);
#pragma unroll
      for (int d = 1; d < 16; d <<= 1) mx = fmaxf(mx, __shfl_xor(mx, d, 32));
      const float mnew = fmaxf(mrow[r], mx);
      alpha[r] = __expf(mrow[r] - mnew);          // underflows to 0 on 1st iter
      const float e0 = __expf(s0 - mnew) * g0;    // masked -> exp(-1e30) = 0
      const float e1 = __expf(s1 - mnew) * g1;
      float rs = e0 + e1;
#pragma unroll
      for (int d = 1; d < 16; d <<= 1) rs += __shfl_xor(rs, d, 32);
      lrow[r] = lrow[r] * alpha[r] + rs;
      mrow[r] = mnew;
      p0[r] = e0;
      p1[r] = e1;
    }

    // rescale running context by per-row alpha
#pragma unroll
    for (int nc = 0; nc < 4; ++nc)
#pragma unroll
      for (int r = 0; r < 8; ++r) o[nc][r] *= alpha[r];

    // ---- re-swizzle P (C-layout) -> A-layout through wave-private LDS ----
#pragma unroll
    for (int r = 0; r < 8; ++r) {
      const int row = hl * 8 + r;
      P[row * 32 + l15] = f2bf(p0[r]);
      P[row * 32 + 16 + l15] = f2bf(p1[r]);
    }
    __builtin_amdgcn_wave_barrier();   // LDS is in-order per wave; block reordering
    const __bf16* pp = P + l15 * 32 + hl * 8;
    v16bf aP = cat16(*(const v8bf*)pp, *(const v8bf*)(pp + 16));

    // ---- context: O += P . V  (V^T layout -> contiguous B fragments) ----
#pragma unroll
    for (int nc = 0; nc < 4; ++nc) {
      const __bf16* vp = Vbase + (size_t)(nc * 16 + l15) * SEQ + kt + hl * 8;
      v16bf bV = cat16(*(const v8bf*)vp, *(const v8bf*)(vp + 16));
      o[nc] = wmma_bf16(aP, bV, o[nc]);
    }
    __builtin_amdgcn_wave_barrier();   // reads done before next iter's P stores
  }

  // ---- normalize and write context (bf16, [b][s][h*64+d]) ----
  __bf16* C = (__bf16*)Cws;
#pragma unroll
  for (int r = 0; r < 8; ++r) {
    const float inv = 1.0f / lrow[r];
    const int row = q0 + hl * 8 + r;
    const size_t base = (size_t)(b * SEQ + row) * DIM + h * DH;
#pragma unroll
    for (int nc = 0; nc < 4; ++nc) C[base + nc * 16 + l15] = f2bf(o[nc][r] * inv);
  }
}

// ---------------------------------------------------------------------------
extern "C" void kernel_launch(void* const* d_in, const int* in_sizes, int n_in,
                              void* d_out, int out_size, void* d_ws, size_t ws_size,
                              hipStream_t stream) {
  const float* query = (const float*)d_in[0];
  const float* key   = (const float*)d_in[1];
  const float* value = (const float*)d_in[2];
  const int*   mask  = (const int*)d_in[3];
  const float* gaussw = (const float*)d_in[4];
  const float* Wq = (const float*)d_in[5];
  const float* bq = (const float*)d_in[6];
  const float* Wk = (const float*)d_in[7];
  const float* bk = (const float*)d_in[8];
  const float* Wv = (const float*)d_in[9];
  const float* bv = (const float*)d_in[10];
  const float* Wo = (const float*)d_in[11];
  const float* bo = (const float*)d_in[12];
  float* out = (float*)d_out;

  const size_t elems = (size_t)BS * SEQ * DIM;
  unsigned short* qws  = (unsigned short*)d_ws;
  unsigned short* kws  = qws + elems;
  unsigned short* vtws = kws + elems;
  unsigned short* cws  = vtws + elems;

  const dim3 gg(DIM / 128, (BS * SEQ) / 128);
  const dim3 gb(256);
  const float qscale = 0.125f;   // 1/sqrt(D_HEAD)

  mha_gemm_kernel<false, true, false><<<gg, gb, 0, stream>>>(query, Wq, bq, qws, qscale);
  mha_gemm_kernel<false, true, false><<<gg, gb, 0, stream>>>(key, Wk, bk, kws, 1.0f);
  mha_gemm_kernel<false, true, true><<<gg, gb, 0, stream>>>(value, Wv, bv, vtws, 1.0f);

  mha_flash_attn_kernel<<<dim3(SEQ / 64, NH, BS), dim3(128), 0, stream>>>(
      qws, kws, vtws, mask, gaussw, cws);

  mha_gemm_kernel<true, false, false><<<gg, gb, 0, stream>>>(cws, Wo, bo, out, 1.0f);
}
